// CausalSelfAttention_14130442404309
// MI455X (gfx1250) — compile-verified, matrix operand order checked by reading, not run on hardware
//
#include <hip/hip_runtime.h>
#include <stdint.h>

typedef __attribute__((ext_vector_type(16))) __bf16        v16bf;
typedef __attribute__((ext_vector_type(8)))  float         v8f;
typedef __attribute__((ext_vector_type(4)))  unsigned int  u32x4;
typedef int i32x4v __attribute__((vector_size(16)));       // matches builtin param

#define HN   16
#define TDIM 2048
#define BDIM 4
#define DDIM 64
#define CDIM 1024

__device__ __forceinline__ unsigned short f2bf(float f) {
  union { float f; unsigned u; } v; v.f = f;
  unsigned u = v.u + 0x7FFFu + ((v.u >> 16) & 1u);
  return (unsigned short)(u >> 16);
}

union FragAB { v16bf bf; u32x4 q[2]; };

__device__ __forceinline__ v8f wmma_bf16(const FragAB& a, const FragAB& b, v8f c) {
  return __builtin_amdgcn_wmma_f32_16x16x32_bf16(false, a.bf, false, b.bf,
                                                 (short)0, c, false, false);
}

__device__ __forceinline__ float redmax16(float v) {
  #pragma unroll
  for (int m = 1; m < 16; m <<= 1) v = fmaxf(v, __shfl_xor(v, m, 32));
  return v;
}
__device__ __forceinline__ float redsum16(float v) {
  #pragma unroll
  for (int m = 1; m < 16; m <<= 1) v += __shfl_xor(v, m, 32);
  return v;
}

// ---- async global->LDS 16B copy (CDNA5 GLOBAL_LOAD_ASYNC_TO_LDS_B128) ------
#if __has_builtin(__builtin_amdgcn_global_load_async_to_lds_b128)
#define HAS_ASYNC_LDS 1
#else
#define HAS_ASYNC_LDS 0
#endif

__device__ __forceinline__ void cp16(unsigned short* dst, const unsigned short* src) {
#if HAS_ASYNC_LDS
  __builtin_amdgcn_global_load_async_to_lds_b128(
      (__attribute__((address_space(1))) i32x4v*)(uintptr_t)src,
      (__attribute__((address_space(3))) i32x4v*)(unsigned)(uintptr_t)dst,
      0, 0);
#else
  *(u32x4*)dst = *(const u32x4*)src;
#endif
}

__device__ __forceinline__ void wait_async() {
#if HAS_ASYNC_LDS
#if __has_builtin(__builtin_amdgcn_s_wait_asynccnt)
  __builtin_amdgcn_s_wait_asynccnt(0);
#else
  asm volatile("s_wait_asynccnt 0x0" ::: "memory");
#endif
#endif
}

// ---------------------------------------------------------------------------
// One-shot pre-convert: x -> bf16, W_attn/W_proj -> bf16 TRANSPOSED [n][k].
// ~13M elements; bandwidth-trivial at 23.3 TB/s, keeps GEMM staging pure copies.
// ---------------------------------------------------------------------------
__global__ __launch_bounds__(256)
void preconvert(const float* __restrict__ x,
                const float* __restrict__ W_attn,
                const float* __restrict__ W_proj,
                unsigned short* __restrict__ xb,
                unsigned short* __restrict__ WtA,
                unsigned short* __restrict__ WtP)
{
  const size_t i = (size_t)blockIdx.x * 256 + threadIdx.x;
  const size_t NX = (size_t)BDIM * TDIM * CDIM;        // 8M
  const size_t NA = (size_t)3 * CDIM * CDIM;           // 3M
  const size_t NP = (size_t)CDIM * CDIM;               // 1M
  if (i < NX) xb[i] = f2bf(x[i]);
  if (i < NA) {
    const size_t n = i >> 10, k = i & 1023;
    WtA[i] = f2bf(W_attn[k * (3 * CDIM) + n]);
  }
  if (i < NP) {
    const size_t n = i >> 10, k = i & 1023;
    WtP[i] = f2bf(W_proj[k * CDIM + n]);
  }
}

// ---------------------------------------------------------------------------
// Tiled bf16 WMMA GEMM, double-buffered async-LDS pipeline.
//   MODE 0: qkv = xb @ WtA^T -> scatter bf16 Q/K/V (Q scaled by 1/sqrt(D))
//   MODE 1: y   = attn @ WtP^T -> fp32 out (+bias, pad-zero)
// Block: 256 threads (8 waves), tile 64(M) x 128(N), K-step 32.
// ---------------------------------------------------------------------------
template<int MODE, int NDIM>
__global__ __launch_bounds__(256)
void gemm_wmma(const unsigned short* __restrict__ Ab,   // MODE0: xb [M][K]; MODE1: Yb [B,H,T,D]
               const unsigned short* __restrict__ Wt,   // [NDIM][CDIM] bf16 (transposed)
               const float* __restrict__ bias,
               unsigned short* __restrict__ Qb,
               unsigned short* __restrict__ Kb,
               unsigned short* __restrict__ Vb,
               const unsigned char* __restrict__ pad,
               float* __restrict__ out)
{
  __shared__ unsigned short sA[2][64][40];    // [m][k], 80B rows (16B aligned)
  __shared__ unsigned short sB[2][128][40];   // [n][k]

  const int tid  = threadIdx.x;
  const int lane = tid & 31, wave = tid >> 5;
  const int wm = wave >> 2, wn = wave & 3;    // 2x4 wave grid
  const int hf = lane >> 4, l16 = lane & 15;
  const int m0 = blockIdx.x * 64, n0 = blockIdx.y * 128;

  auto stageA = [&](int buf, int k0) {
    const int row = tid >> 2, koff = (tid & 3) * 8;
    const int m = m0 + row;
    const unsigned short* src;
    if constexpr (MODE == 0) {
      src = Ab + (size_t)m * CDIM + k0 + koff;
    } else {
      const int kg = k0 + koff;
      const int hh = kg >> 6, dd = kg & 63;
      const int b = m >> 11, t = m & (TDIM - 1);
      src = Ab + (((size_t)b * HN + hh) * TDIM + t) * DDIM + dd;
    }
    cp16(&sA[buf][row][koff], src);
  };
  auto stageB = [&](int buf, int k0) {
    #pragma unroll
    for (int i = 0; i < 2; i++) {
      const int c = tid * 2 + i;              // 0..511 chunks of 8 bf16
      const int row = c >> 2, koff = (c & 3) * 8;
      cp16(&sB[buf][row][koff], Wt + (size_t)(n0 + row) * CDIM + k0 + koff);
    }
  };

  const v8f vz = {0.f,0.f,0.f,0.f,0.f,0.f,0.f,0.f};
  v8f acc[2][2] = {{vz, vz}, {vz, vz}};

  stageA(0, 0); stageB(0, 0);
  int cur = 0;
  for (int k0 = 0; k0 < CDIM; k0 += 32) {
    wait_async();
    __syncthreads();                          // buf[cur] ready on all waves
    if (k0 + 32 < CDIM) { stageA(cur ^ 1, k0 + 32); stageB(cur ^ 1, k0 + 32); }

    FragAB af[2], bf2[2];
    #pragma unroll
    for (int sm = 0; sm < 2; sm++) {
      const unsigned short* p = &sA[cur][wm * 32 + sm * 16 + l16][hf * 8];
      af[sm].q[0] = *(const u32x4*)p;
      af[sm].q[1] = *(const u32x4*)(p + 16);
    }
    #pragma unroll
    for (int sn = 0; sn < 2; sn++) {
      const unsigned short* p = &sB[cur][wn * 32 + sn * 16 + l16][hf * 8];
      bf2[sn].q[0] = *(const u32x4*)p;
      bf2[sn].q[1] = *(const u32x4*)(p + 16);
    }
    #pragma unroll
    for (int sm = 0; sm < 2; sm++)
      #pragma unroll
      for (int sn = 0; sn < 2; sn++)
        acc[sm][sn] = wmma_bf16(af[sm], bf2[sn], acc[sm][sn]);
    cur ^= 1;
  }

  // ---- epilogue ----
  #pragma unroll
  for (int sm = 0; sm < 2; sm++) {
    #pragma unroll
    for (int sn = 0; sn < 2; sn++) {
      #pragma unroll
      for (int r = 0; r < 8; r++) {
        const int m = m0 + wm * 32 + sm * 16 + hf * 8 + r;
        const int n = n0 + wn * 32 + sn * 16 + l16;
        const float val = acc[sm][sn][r] + bias[n];
        const int b = m >> 11, t = m & (TDIM - 1);
        if constexpr (MODE == 0) {
          const int which = n >> 10, rem = n & 1023;
          const int hh = rem >> 6, dd = rem & 63;
          const size_t idx = (((size_t)b * HN + hh) * TDIM + t) * DDIM + dd;
          if (which == 0)      Qb[idx] = f2bf(val * 0.125f);  // fold 1/sqrt(64)
          else if (which == 1) Kb[idx] = f2bf(val);
          else                 Vb[idx] = f2bf(val);
        } else {
          out[(size_t)m * CDIM + n] = pad[b * TDIM + t] ? 0.0f : val;
        }
      }
    }
  }
}

// ---------------------------------------------------------------------------
// Flash attention: block = (b,h) x 128-query tile, 8 waves of 16 rows each.
// Double-buffered K/V tiles (K async-to-LDS), online softmax fp32, bf16 WMMA.
// ---------------------------------------------------------------------------
__global__ __launch_bounds__(256)
void flash_attn(const unsigned short* __restrict__ Qb,
                const unsigned short* __restrict__ Kb,
                const unsigned short* __restrict__ Vb,
                const unsigned char* __restrict__ pad,
                unsigned short* __restrict__ Yb)
{
  __shared__ unsigned short sK[2][64][72];     // [key][d] (== B^T layout for Q.K^T)
  __shared__ unsigned short sV[2][64][72];     // [d][key] (transposed V for P.V)
  __shared__ unsigned short sP[8][16][72];     // per-wave P staging (C->A relayout)

  const int tid  = threadIdx.x;
  const int lane = tid & 31, wave = tid >> 5;
  const int hf = lane >> 4, l16 = lane & 15;
  const int bh = blockIdx.y;                   // b*H + h
  const int b  = bh >> 4;
  const int q0 = blockIdx.x * 128;
  const size_t base = (size_t)bh * TDIM * DDIM;

  auto stage_kv = [&](int buf, int ks) {
    const int row = tid >> 2, doff = (tid & 3) * 16;
    const unsigned short* kp = Kb + base + (size_t)(ks + row) * DDIM + doff;
    cp16(&sK[buf][row][doff],     kp);
    cp16(&sK[buf][row][doff + 8], kp + 8);
    const unsigned short* vp = Vb + base + (size_t)(ks + row) * DDIM + doff;
    u32x4 v0 = *(const u32x4*)vp;
    u32x4 v1 = *(const u32x4*)(vp + 8);
    #pragma unroll
    for (int e = 0; e < 4; e++) {
      unsigned w0 = v0[e], w1 = v1[e];
      sV[buf][doff + 2 * e    ][row] = (unsigned short)(w0 & 0xFFFFu);
      sV[buf][doff + 2 * e + 1][row] = (unsigned short)(w0 >> 16);
      sV[buf][doff + 8 + 2 * e    ][row] = (unsigned short)(w1 & 0xFFFFu);
      sV[buf][doff + 8 + 2 * e + 1][row] = (unsigned short)(w1 >> 16);
    }
  };

  // Q fragments for this wave's 16 rows (held across the key loop)
  FragAB qf[2];
  {
    const unsigned short* qp = Qb + base + (size_t)(q0 + wave * 16 + l16) * DDIM;
    #pragma unroll
    for (int c = 0; c < 2; c++) {
      qf[c].q[0] = *(const u32x4*)(qp + c * 32 + hf * 8);
      qf[c].q[1] = *(const u32x4*)(qp + c * 32 + hf * 8 + 16);
    }
  }

  const v8f vz = {0.f,0.f,0.f,0.f,0.f,0.f,0.f,0.f};
  v8f O[4] = {vz, vz, vz, vz};
  float mrun[8], lrun[8];
  #pragma unroll
  for (int r = 0; r < 8; r++) { mrun[r] = -INFINITY; lrun[r] = 0.f; }

  const int nkb = (q0 >> 6) + 2;               // causal limit: key blocks of 64
  stage_kv(0, 0);
  int cur = 0;
  for (int kb = 0; kb < nkb; ++kb) {
    wait_async();
    __syncthreads();                           // buf[cur] ready on all waves
    if (kb + 1 < nkb) stage_kv(cur ^ 1, (kb + 1) * 64);
    const int ks = kb * 64;

    // ---- S = Q @ K^T  (16q x 64k per wave) ----
    v8f S[4];
    #pragma unroll
    for (int n = 0; n < 4; n++) {
      v8f a = vz;
      #pragma unroll
      for (int c = 0; c < 2; c++) {
        FragAB kf;
        const unsigned short* p = &sK[cur][n * 16 + l16][c * 32 + hf * 8];
        kf.q[0] = *(const u32x4*)p;
        kf.q[1] = *(const u32x4*)(p + 16);
        a = wmma_bf16(qf[c], kf, a);
      }
      S[n] = a;
    }

    // ---- causal + padding mask, row-max ----
    float rmax[8];
    #pragma unroll
    for (int r = 0; r < 8; r++) rmax[r] = -INFINITY;
    #pragma unroll
    for (int n = 0; n < 4; n++) {
      const int kg = ks + n * 16 + l16;
      const bool kvalid = (pad[b * TDIM + kg] == 0);
      #pragma unroll
      for (int r = 0; r < 8; r++) {
        const int qg = q0 + wave * 16 + hf * 8 + r;
        const float s = (kvalid && kg <= qg) ? S[n][r] : -INFINITY;
        S[n][r] = s;
        rmax[r] = fmaxf(rmax[r], s);
      }
    }

    // ---- online softmax update ----
    #pragma unroll
    for (int r = 0; r < 8; r++) {
      const float rm = redmax16(rmax[r]);
      const float mnew = fmaxf(mrun[r], rm);
      float alpha, psum = 0.f;
      if (mnew == -INFINITY) {                 // row fully masked so far
        alpha = 1.f;
        #pragma unroll
        for (int n = 0; n < 4; n++) S[n][r] = 0.f;
      } else {
        alpha = __expf(mrun[r] - mnew);        // -inf -> 0, finite -> rescale
        #pragma unroll
        for (int n = 0; n < 4; n++) {
          const float p = __expf(S[n][r] - mnew);
          S[n][r] = p;
          psum += p;
        }
      }
      psum = redsum16(psum);
      mrun[r] = mnew;
      lrun[r] = lrun[r] * alpha + psum;
      #pragma unroll
      for (int j = 0; j < 4; j++) O[j][r] *= alpha;
    }

    // ---- P: C-layout -> A-layout via wave-private LDS (same-wave LDS is
    //      in-order, no barrier needed) ----
    #pragma unroll
    for (int n = 0; n < 4; n++)
      #pragma unroll
      for (int r = 0; r < 8; r++)
        sP[wave][hf * 8 + r][n * 16 + l16] = f2bf(S[n][r]);

    FragAB pf[2];
    #pragma unroll
    for (int c = 0; c < 2; c++) {
      const unsigned short* p = &sP[wave][l16][c * 32 + hf * 8];
      pf[c].q[0] = *(const u32x4*)p;
      pf[c].q[1] = *(const u32x4*)(p + 16);
    }
    // ---- O += P @ V ----
    #pragma unroll
    for (int j = 0; j < 4; j++) {
      #pragma unroll
      for (int c = 0; c < 2; c++) {
        FragAB vf;
        const unsigned short* p = &sV[cur][j * 16 + l16][c * 32 + hf * 8];
        vf.q[0] = *(const u32x4*)p;
        vf.q[1] = *(const u32x4*)(p + 16);
        O[j] = wmma_bf16(pf[c], vf, O[j]);
      }
    }
    cur ^= 1;
  }

  // ---- normalize and write bf16 [B,H,T,D] ----
  #pragma unroll
  for (int j = 0; j < 4; j++) {
    #pragma unroll
    for (int r = 0; r < 8; r++) {
      const int q = q0 + wave * 16 + hf * 8 + r;
      const float l = lrun[r];
      const float o = (l > 0.f) ? O[j][r] / l : 0.f;
      Yb[base + (size_t)q * DDIM + j * 16 + l16] = f2bf(o);
    }
  }
}

extern "C" void kernel_launch(void* const* d_in, const int* in_sizes, int n_in,
                              void* d_out, int out_size, void* d_ws, size_t ws_size,
                              hipStream_t stream) {
  (void)in_sizes; (void)n_in; (void)out_size; (void)ws_size;
  const float*         x      = (const float*)d_in[0];
  const unsigned char* pad    = (const unsigned char*)d_in[1];  // bool [B,T]
  const float*         W_attn = (const float*)d_in[2];
  const float*         b_attn = (const float*)d_in[3];
  const float*         W_proj = (const float*)d_in[4];
  const float*         b_proj = (const float*)d_in[5];
  float*               out    = (float*)d_out;

  const size_t ebhtd = (size_t)BDIM * HN * TDIM * DDIM;   // 8M
  unsigned short* Qb  = (unsigned short*)d_ws;
  unsigned short* Kb  = Qb  + ebhtd;
  unsigned short* Vb  = Kb  + ebhtd;
  unsigned short* Yb  = Vb  + ebhtd;
  unsigned short* xb  = Yb  + ebhtd;                      // 8M
  unsigned short* WtA = xb  + (size_t)BDIM * TDIM * CDIM; // 3M
  unsigned short* WtP = WtA + (size_t)3 * CDIM * CDIM;    // 1M

  dim3 blk(256);
  preconvert<<<dim3(32768), blk, 0, stream>>>(x, W_attn, W_proj, xb, WtA, WtP);
  // QKV projection: M=8192, N=3072, K=1024
  gemm_wmma<0, 3 * CDIM><<<dim3(128, 24), blk, 0, stream>>>(
      xb, WtA, b_attn, Qb, Kb, Vb, nullptr, nullptr);
  // Flash attention: 16 q-tiles x 64 (b,h) pairs
  flash_attn<<<dim3(TDIM / 128, BDIM * HN), blk, 0, stream>>>(Qb, Kb, Vb, pad, Yb);
  // Output projection: M=8192, N=1024, K=1024
  gemm_wmma<1, CDIM><<<dim3(128, 8), blk, 0, stream>>>(
      Yb, WtP, b_proj, nullptr, nullptr, nullptr, pad, out);
}
